// OrbitWarsGCN_53102975648306
// MI455X (gfx1250) — matrix operand order repaired and checked by reading, not compiled, not run on hardware
//
#include <hip/hip_runtime.h>
#include <hip/hip_bf16.h>
#include <math.h>

// ---------------------------------------------------------------------------
// OrbitWarsGCN for MI455X (gfx1250, wave32, WMMA).
//
// B=4, N=384, H=64, K=4. The pair head's concat input factorizes across the
// first matmul: hidden = relu(A_i + B_j + C_b + dist*U + rv*V), collapsing the
// 12.6 GFLOP pair GEMM to two [1536,64]x[64,64] projections + streaming work.
// C_b is folded into the A projection as a per-batch bias. ew is kept in f16.
// Pair blocks keep B_j/w2/U/V resident in LDS across 16 i's -> global reads
// shrink to ~unique data; the 9.4 MB masked logits write is the roofline
// floor (~0.4 us @ 23.3 TB/s). All matrix ops are v_wmma_f32_16x16x32_f16;
// B-operand tiles staged transposed in LDS so fragments load as ds_load_b128.
// ---------------------------------------------------------------------------

#define Bsz 4
#define Nn  384
#define Hh  64
#define NNK (Nn * Nn * 4)          // 589824
#define LOGITS_PER_B (NNK + 1)     // 589825

typedef __attribute__((ext_vector_type(16))) _Float16 v16h;
typedef __attribute__((ext_vector_type(8)))  float    v8f;

// ---- WMMA fragment loaders (this file's contiguous k order: K = 16*(lane/16) + elem) -----

// A-matrix 16x32 f16 from row-major LDS tile buf[m][k] (ld halves/row):
// lanes 0-15 -> M=lane, elems 0..15 = K 0..15
// lanes 16-31 -> M=lane-16, elems 0..15 = K 16..31
// Each 8-elem run is contiguous -> ds_load_b128.
__device__ __forceinline__ v16h load_a16(const _Float16* buf, int ld, int kbase) {
  int lane = threadIdx.x & 31;
  int m = lane & 15;
  int khi = ((lane >> 4) & 1) * 16;
  v16h a;
#pragma unroll
  for (int e = 0; e < 16; ++e) {
    int k = kbase + e + khi;
    a[e] = buf[m * ld + k];
  }
  return a;
}

// B-matrix 32x16 f16 from TRANSPOSED LDS tile buf[n][k] (ld halves/row):
// lane -> N = lane%16; lanes 0-15 hold K k0..k0+15, lanes 16-31 K k0+16..k0+31.
// 16 contiguous halves per lane -> 2x ds_load_b128.
__device__ __forceinline__ v16h load_bT16(const _Float16* buf, int ld, int kbase) {
  int lane = threadIdx.x & 31;
  int n = lane & 15;
  int k0 = kbase + ((lane >> 4) & 1) * 16;
  v16h b;
#pragma unroll
  for (int e = 0; e < 16; ++e) b[e] = buf[n * ld + k0 + e];
  return b;
}

#define WMMA_F16(a, b, c) \
  __builtin_amdgcn_wmma_f32_16x16x32_f16(false, (a), false, (b), (short)0, (c), false, false)

// ---------------------------------------------------------------------------
// Prep: g = relu(global@ge_w + ge_b); C = g@W_g + ah_b1; U/V edge-row folds.
// ---------------------------------------------------------------------------
__global__ void prep_kernel(const float* __restrict__ gf, const float* __restrict__ ge_w,
                            const float* __restrict__ ge_b, const float* __restrict__ ah_w1,
                            const float* __restrict__ ah_b1,
                            float* __restrict__ g, float* __restrict__ C,
                            float* __restrict__ U, float* __restrict__ V) {
  __shared__ float gsh[Bsz * 32];
  int tid = threadIdx.x;  // 256
  if (tid < Bsz * 32) {
    int b = tid >> 5, q = tid & 31;
    float a = ge_b[q];
#pragma unroll
    for (int k = 0; k < 6; ++k) a += gf[b * 6 + k] * ge_w[k * 32 + q];
    a = fmaxf(a, 0.f);
    gsh[tid] = a;
    g[tid] = a;
  }
  if (tid < Hh) {
    U[tid] = ah_w1[160 * Hh + tid] + 0.5f * ah_w1[162 * Hh + tid];
    V[tid] = ah_w1[161 * Hh + tid];
  }
  __syncthreads();
  {
    int b = tid >> 6, d = tid & 63;
    float a = ah_b1[d];
#pragma unroll
    for (int q = 0; q < 32; ++q) a += gsh[b * 32 + q] * ah_w1[(128 + q) * Hh + d];
    C[b * Hh + d] = a;
  }
}

// ---------------------------------------------------------------------------
// Encoder layer 1 (K=12, tiny): h1 = relu(X @ ne_w1 + b1). One thread/elem.
// ---------------------------------------------------------------------------
__global__ void enc1_kernel(const float* __restrict__ X, const float* __restrict__ w1,
                            const float* __restrict__ b1, float* __restrict__ Y) {
  int idx = blockIdx.x * 256 + threadIdx.x;   // < 1536*64
  int d = idx & 63, row = idx >> 6;
  float a = b1[d];
#pragma unroll
  for (int k = 0; k < 12; ++k) a += X[row * 12 + k] * w1[k * Hh + d];
  Y[idx] = fmaxf(a, 0.f);
}

// ---------------------------------------------------------------------------
// WMMA GEMM, compile-time specialized: Y = act((X [+X2]) @ W64x64 [+bias]).
// BIAS_MODE: 0 = none, 1 = per-output bias[n], 2 = per-batch bias[b*64+n]
// (used to fold C_b into the A projection). One wave per 16x16 output tile.
// ---------------------------------------------------------------------------
#define WPAD 72   // padded row stride (halves) for transposed B tiles

template <bool HAS_X2, int BIAS_MODE, bool RELU>
__global__ void gemm64_wmma(const float* __restrict__ X, const float* __restrict__ X2,
                            const float* __restrict__ W, const float* __restrict__ bias,
                            float* __restrict__ Y) {
  __shared__ _Float16 xs[16][Hh];       // A tile, row-major
  __shared__ _Float16 wT[16][WPAD];     // B tile, transposed [n][k]
  int lane = threadIdx.x;
  int row0 = blockIdx.x * 16;
  int n0 = blockIdx.y * 16;
#pragma unroll
  for (int t = lane; t < 16 * Hh; t += 32) {
    int r = t >> 6, c = t & 63;
    float v = X[(row0 + r) * Hh + c];
    if (HAS_X2) v += X2[(row0 + r) * Hh + c];
    xs[r][c] = (_Float16)v;
  }
#pragma unroll
  for (int t = lane; t < Hh * 16; t += 32) {
    int k = t >> 4, n = t & 15;         // consecutive lanes -> consecutive n (coalesced)
    wT[n][k] = (_Float16)W[k * Hh + n0 + n];
  }
  __syncthreads();
  v16h a0 = load_a16(&xs[0][0], Hh, 0);
  v16h a1 = load_a16(&xs[0][0], Hh, 32);
  v16h b0 = load_bT16(&wT[0][0], WPAD, 0);
  v16h b1 = load_bT16(&wT[0][0], WPAD, 32);
  v8f c = {};
  c = WMMA_F16(a0, b0, c);
  c = WMMA_F16(a1, b1, c);
  int n = lane & 15, hi = lane >> 4;
  float bv = 0.f;
  if (BIAS_MODE == 1) bv = bias[n0 + n];
  if (BIAS_MODE == 2) bv = bias[(row0 / Nn) * Hh + n0 + n];   // tiles never straddle batches
#pragma unroll
  for (int r = 0; r < 8; ++r) {
    int m = r + hi * 8;
    float v = c[r] + bv;
    if (RELU) v = fmaxf(v, 0.f);
    Y[(row0 + m) * Hh + n0 + n] = v;
  }
}

// ---------------------------------------------------------------------------
// Edge MLP: ew[b,i,j] = sigmoid(relu(e@en_w1+b1)@en_w2+b2), stored f16
// (range [0,1] -> f16 safe; halves the biggest intermediate's traffic).
// ---------------------------------------------------------------------------
__global__ void edge_kernel(const float* __restrict__ pos, const float* __restrict__ vel,
                            const float* __restrict__ en_w1, const float* __restrict__ en_b1,
                            const float* __restrict__ en_w2, const float* __restrict__ en_b2,
                            _Float16* __restrict__ ew) {
  int idx = blockIdx.x * 256 + threadIdx.x;   // < B*N*N
  int j = idx % Nn;
  int r = idx / Nn;
  int i = r % Nn;
  int b = r / Nn;
  int bi = b * Nn + i, bj = b * Nn + j;
  float dx = pos[bj * 2]     - pos[bi * 2];
  float dy = pos[bj * 2 + 1] - pos[bi * 2 + 1];
  float sq = dx * dx + dy * dy;
  float dist = sq > 0.f ? sqrtf(sq) : 0.f;
  float wx = vel[bj * 2]     - vel[bi * 2];
  float wy = vel[bj * 2 + 1] - vel[bi * 2 + 1];
  float sv = wx * wx + wy * wy;
  float rv = sv > 0.f ? sqrtf(sv) : 0.f;
  float e2 = dist * 0.5f;
  float s = en_b2[0];
#pragma unroll
  for (int q = 0; q < 16; ++q) {
    float t = en_w1[q] * dist + en_w1[16 + q] * rv + en_w1[32 + q] * e2 + en_b1[q];
    s += fmaxf(t, 0.f) * en_w2[q];
  }
  ew[idx] = (_Float16)(1.f / (1.f + __expf(-s)));
}

// ---------------------------------------------------------------------------
// GCN messages: msgs[b,i,:] = sum_j ew[b,i,j] * h[b,j,:]. WMMA, K=384 in 12
// chunks of 32. ew is already f16 -> A tile staged with u64 copies
// (global_load_b64 + ds_store_b64, no conversion). h chunk staged transposed.
// ---------------------------------------------------------------------------
#define HPAD 40

__global__ void msg_wmma(const _Float16* __restrict__ ew, const float* __restrict__ h,
                         float* __restrict__ msgs) {
  __shared__ __align__(16) _Float16 es[16][32];   // A: ew tile, row-major
  __shared__ _Float16 hT[16][HPAD];               // B: h chunk, transposed [n][k]
  int lane = threadIdx.x;
  int bt = blockIdx.x;                  // b*(N/16) + mtile
  int b = bt / (Nn / 16), mt = bt % (Nn / 16);
  int n0 = blockIdx.y * 16;
  int i0 = mt * 16;
  v8f c = {};
  for (int kc = 0; kc < Nn; kc += 32) {
    __syncthreads();                    // protect LDS reuse across iterations
    // A tile: 16 rows x 16 halves = 64 u64 words; 2 per lane, type-preserving
    {
      const unsigned long long* src =
          (const unsigned long long*)(ew + (size_t)(b * Nn + i0) * Nn + kc);
      unsigned long long* dst = (unsigned long long*)(&es[0][0]);
#pragma unroll
      for (int q = lane; q < 64; q += 32) {
        int r = q >> 2, c8 = q & 3;
        dst[r * 4 + c8] = src[(size_t)r * (Nn / 4) + c8];
      }
    }
#pragma unroll
    for (int t = lane; t < 32 * 16; t += 32) {
      int kk = t >> 4, n = t & 15;      // consecutive lanes -> consecutive n
      hT[n][kk] = (_Float16)h[(b * Nn + kc + kk) * Hh + n0 + n];
    }
    __syncthreads();
    v16h a0 = load_a16(&es[0][0], 32, 0);
    v16h b0 = load_bT16(&hT[0][0], HPAD, 0);
    c = WMMA_F16(a0, b0, c);
  }
  int n = lane & 15, hi = lane >> 4;
#pragma unroll
  for (int r = 0; r < 8; ++r)
    msgs[(b * Nn + i0 + r + hi * 8) * Hh + n0 + n] = c[r];
}

// ---------------------------------------------------------------------------
// Pair head. Block (256 thr, 8 waves) = one (b, j-tile, 16-i chunk):
//   stage B_j tile / w2 / U / V into LDS once, loop over i (2 per wave):
//     hidden = relu(A'_i + B_j + dist*U + rv*V)   (A' already contains C_b)
//     logits = hidden @ W2pad -> 2 WMMAs -> 64 contiguous masked stores.
// ---------------------------------------------------------------------------
#define ITILE 16

__global__ void pair_wmma(const float* __restrict__ A, const float* __restrict__ Bm,
                          const float* __restrict__ U, const float* __restrict__ V,
                          const float* __restrict__ pos, const float* __restrict__ vel,
                          const float* __restrict__ ah_w2, const float* __restrict__ ah_b2,
                          const unsigned char* __restrict__ amask,
                          float* __restrict__ out) {
  __shared__ float Bsh[16][Hh];               // B_j tile (f32)
  __shared__ _Float16 w2T[16][WPAD];          // padded second-layer weights [n][k]
  __shared__ float Ush[Hh], Vsh[Hh];
  __shared__ _Float16 hid[8][16][Hh];         // per-wave hidden tiles
  __shared__ float dt[8][16][4];              // per-wave logits tiles
  int tid = threadIdx.x, lane = tid & 31, wave = tid >> 5;
  int id = blockIdx.x;                        // B * (N/16) * (N/ITILE)
  int jt = id % (Nn / 16); id /= (Nn / 16);
  int ic = id % (Nn / ITILE);
  int b = id / (Nn / ITILE);
  int j0 = jt * 16, ibase = ic * ITILE;

  // ---- block-wide staging (done once) ----
  for (int t = tid; t < 16 * Hh; t += 256) {
    int r = t >> 6, c = t & 63;
    Bsh[r][c] = Bm[(size_t)(b * Nn + j0 + r) * Hh + c];
  }
  for (int t = tid; t < Hh * 16; t += 256) {
    int k = t >> 4, n = t & 15;
    w2T[n][k] = (n < 4) ? (_Float16)ah_w2[k * 4 + n] : (_Float16)0.f;
  }
  if (tid < Hh) { Ush[tid] = U[tid]; Vsh[tid] = V[tid]; }
  __syncthreads();

  // ---- per-lane constants, invariant across the i loop ----
  int jl = lane >> 1;                         // j_local 0..15
  int hb = (lane & 1) * 32;                   // hidden-dim half
  int bj = b * Nn + j0 + jl;
  float pjx = pos[bj * 2], pjy = pos[bj * 2 + 1];
  float vjx = vel[bj * 2], vjy = vel[bj * 2 + 1];
  int n = lane & 15, hi = lane >> 4;
  float bv = (n < 4) ? ah_b2[n] : 0.f;
  v16h b0 = load_bT16(&w2T[0][0], WPAD, 0);
  v16h b1 = load_bT16(&w2T[0][0], WPAD, 32);

#pragma unroll
  for (int iw = 0; iw < ITILE / 8; ++iw) {    // 2 iterations, 8 waves cover 8 i's each
    int i = ibase + iw * 8 + wave;
    int bi = b * Nn + i;
    const float* Ai = A + (size_t)bi * Hh + hb;
    __builtin_prefetch(Ai, 0, 3);             // global_prefetch_b8 (probe-confirmed)
    float pix = pos[bi * 2], piy = pos[bi * 2 + 1];
    float vix = vel[bi * 2], viy = vel[bi * 2 + 1];
    float dx = pjx - pix, dy = pjy - piy;
    float sq = dx * dx + dy * dy;
    float dist = sq > 0.f ? sqrtf(sq) : 0.f;
    float wx = vjx - vix, wy = vjy - viy;
    float sv = wx * wx + wy * wy;
    float rv = sv > 0.f ? sqrtf(sv) : 0.f;

#pragma unroll
    for (int d = 0; d < 32; ++d) {
      float v = Ai[d] + Bsh[jl][hb + d] + dist * Ush[hb + d] + rv * Vsh[hb + d];
      hid[wave][jl][hb + d] = (_Float16)fmaxf(v, 0.f);
    }
    __syncthreads();

    v16h a0 = load_a16(&hid[wave][0][0], Hh, 0);
    v16h a1 = load_a16(&hid[wave][0][0], Hh, 32);
    v8f c = {};
    c = WMMA_F16(a0, b0, c);
    c = WMMA_F16(a1, b1, c);

    if (n < 4) {
#pragma unroll
      for (int r = 0; r < 8; ++r) dt[wave][r + hi * 8][n] = c[r] + bv;
    }
    __syncthreads();

    long base = (long)b * LOGITS_PER_B + ((long)i * Nn + j0) * 4;
#pragma unroll
    for (int t2 = lane; t2 < 64; t2 += 32) {
      float v = dt[wave][t2 >> 2][t2 & 3];
      out[base + t2] = amask[base + t2] ? v : -1e9f;
    }
  }
}

// ---------------------------------------------------------------------------
// Heads: masked mean pool over nodes, noop logit (masked) + value. 1 block.
// ---------------------------------------------------------------------------
__global__ void heads_kernel(const float* __restrict__ h, const float* __restrict__ valid,
                             const float* __restrict__ g,
                             const float* __restrict__ nh_w1, const float* __restrict__ nh_b1,
                             const float* __restrict__ nh_w2, const float* __restrict__ nh_b2,
                             const float* __restrict__ vh_w1, const float* __restrict__ vh_b1,
                             const float* __restrict__ vh_w2, const float* __restrict__ vh_b2,
                             const unsigned char* __restrict__ amask,
                             float* __restrict__ out) {
  __shared__ float gc[Bsz][96];
  __shared__ float t1[Bsz][32];
  __shared__ float t2[Bsz][64];
  int tid = threadIdx.x;          // 256
  int b = tid >> 6, d = tid & 63;
  float s = 0.f, cnt = 0.f;
  for (int i = 0; i < Nn; ++i) {
    float m = valid[b * Nn + i];
    s += h[(b * Nn + i) * Hh + d] * m;
    cnt += m;
  }
  gc[b][d] = s / fmaxf(cnt, 1.f);
  if (d < 32) gc[b][64 + d] = g[b * 32 + d];
  __syncthreads();
  if (d < 32) {
    float a = nh_b1[d];
    for (int k = 0; k < 96; ++k) a += gc[b][k] * nh_w1[k * 32 + d];
    t1[b][d] = fmaxf(a, 0.f);
  }
  {
    float a = vh_b1[d];
    for (int k = 0; k < 96; ++k) a += gc[b][k] * vh_w1[k * 64 + d];
    t2[b][d] = fmaxf(a, 0.f);
  }
  __syncthreads();
  if (d == 0) {
    float nv = nh_b2[0];
    for (int q = 0; q < 32; ++q) nv += t1[b][q] * nh_w2[q];
    long idx = (long)b * LOGITS_PER_B + NNK;
    out[idx] = amask[idx] ? nv : -1e9f;
    float vv = vh_b2[0];
    for (int q = 0; q < 64; ++q) vv += t2[b][q] * vh_w2[q];
    out[(long)Bsz * LOGITS_PER_B + b] = vv;
  }
}

// ---------------------------------------------------------------------------
extern "C" void kernel_launch(void* const* d_in, const int* in_sizes, int n_in,
                              void* d_out, int out_size, void* d_ws, size_t ws_size,
                              hipStream_t stream) {
  const float* node_features   = (const float*)d_in[0];
  const float* global_features = (const float*)d_in[1];
  const float* positions       = (const float*)d_in[2];
  const float* velocities      = (const float*)d_in[3];
  const float* valid_mask      = (const float*)d_in[4];
  const unsigned char* amask   = (const unsigned char*)d_in[5];
  const float* ne_w1 = (const float*)d_in[6];
  const float* ne_b1 = (const float*)d_in[7];
  const float* ne_w2 = (const float*)d_in[8];
  const float* ne_b2 = (const float*)d_in[9];
  const float* gcn1_w = (const float*)d_in[10];
  const float* gcn1_b = (const float*)d_in[11];
  const float* gcn2_w = (const float*)d_in[12];
  const float* gcn2_b = (const float*)d_in[13];
  const float* en_w1 = (const float*)d_in[14];
  const float* en_b1 = (const float*)d_in[15];
  const float* en_w2 = (const float*)d_in[16];
  const float* en_b2 = (const float*)d_in[17];
  const float* ge_w = (const float*)d_in[18];
  const float* ge_b = (const float*)d_in[19];
  const float* ah_w1 = (const float*)d_in[20];
  const float* ah_b1 = (const float*)d_in[21];
  const float* ah_w2 = (const float*)d_in[22];
  const float* ah_b2 = (const float*)d_in[23];
  const float* nh_w1 = (const float*)d_in[24];
  const float* nh_b1 = (const float*)d_in[25];
  const float* nh_w2 = (const float*)d_in[26];
  const float* nh_b2 = (const float*)d_in[27];
  const float* vh_w1 = (const float*)d_in[28];
  const float* vh_b1 = (const float*)d_in[29];
  const float* vh_w2 = (const float*)d_in[30];
  const float* vh_b2 = (const float*)d_in[31];
  float* out = (float*)d_out;

  // Workspace layout (fp32-element slots; ew slot reused as f16)
  float* ws = (float*)d_ws;
  const size_t NH = (size_t)Bsz * Nn * Hh;     // 98304
  float* hA   = ws;
  float* hB   = hA + NH;
  float* tmp  = hB + NH;
  float* Abuf = tmp + NH;
  float* Bbuf = Abuf + NH;
  float* ewf  = Bbuf + NH;                     // B*N*N slot (f16 uses half of it)
  _Float16* ew16 = (_Float16*)ewf;
  float* gbuf = ewf + (size_t)Bsz * Nn * Nn;
  float* Cbuf = gbuf + Bsz * 32;
  float* Ubuf = Cbuf + Bsz * Hh;
  float* Vbuf = Ubuf + Hh;

  const int MT = (Bsz * Nn) / 16;              // 96 row tiles
  dim3 gemmGrid(MT, Hh / 16);                  // (96,4), 1 wave each
  dim3 msgGrid(Bsz * (Nn / 16), Hh / 16);      // (96,4)

  // 1) per-batch global encodings + pair-head constants
  prep_kernel<<<1, 256, 0, stream>>>(global_features, ge_w, ge_b, ah_w1, ah_b1,
                                     gbuf, Cbuf, Ubuf, Vbuf);
  // 2) node encoder
  enc1_kernel<<<(Bsz * Nn * Hh) / 256, 256, 0, stream>>>(node_features, ne_w1, ne_b1, tmp);
  gemm64_wmma<false, 1, true><<<gemmGrid, 32, 0, stream>>>(tmp, nullptr, ne_w2, ne_b2, hA);
  // 3) edge weights (f16)
  edge_kernel<<<(Bsz * Nn * Nn) / 256, 256, 0, stream>>>(positions, velocities,
                                                         en_w1, en_b1, en_w2, en_b2, ew16);
  // 4) GCN round 1: msgs = ew@h ; h' = relu((h+msgs)@W1+b1)
  msg_wmma<<<msgGrid, 32, 0, stream>>>(ew16, hA, tmp);
  gemm64_wmma<true, 1, true><<<gemmGrid, 32, 0, stream>>>(hA, tmp, gcn1_w, gcn1_b, hB);
  // 5) GCN round 2
  msg_wmma<<<msgGrid, 32, 0, stream>>>(ew16, hB, tmp);
  gemm64_wmma<true, 1, true><<<gemmGrid, 32, 0, stream>>>(hB, tmp, gcn2_w, gcn2_b, hA);
  // 6) factorized projections: A = h@W_hi + C_b (per-batch bias), B = h@W_hj
  gemm64_wmma<false, 2, false><<<gemmGrid, 32, 0, stream>>>(hA, nullptr, ah_w1, Cbuf, Abuf);
  gemm64_wmma<false, 0, false><<<gemmGrid, 32, 0, stream>>>(hA, nullptr, ah_w1 + 64 * Hh,
                                                            nullptr, Bbuf);
  // 7) pair kernel: 2304 blocks x 8 waves, LDS-resident B_j tiles
  pair_wmma<<<Bsz * (Nn / 16) * (Nn / ITILE), 256, 0, stream>>>(Abuf, Bbuf, Ubuf, Vbuf,
                                                                positions, velocities,
                                                                ah_w2, ah_b2, amask, out);
  // 8) noop + value heads
  heads_kernel<<<1, 256, 0, stream>>>(hA, valid_mask, gbuf,
                                      nh_w1, nh_b1, nh_w2, nh_b2,
                                      vh_w1, vh_b1, vh_w2, vh_b2,
                                      amask, out);
}
